// GraphAT_70128226009469
// MI455X (gfx1250) — compile-verified, matrix operand order checked by reading, not run on hardware
//
#include <hip/hip_runtime.h>
#include <hip/hip_bf16.h>
#include <math.h>

#define NV 4096     // voc_size / node count
#define DV 256      // emb dim
#define LRELU_ALPHA 0.2f

typedef __attribute__((ext_vector_type(16))) __bf16 v16bf;
typedef __attribute__((ext_vector_type(8)))  float  v8f;

struct U32x8 { uint4 lo, hi; };   // 32B carrier for a v16bf fragment

// --- CDNA5 async global->LDS DMA (GLOBAL_LOAD_ASYNC_TO_LDS_B128, ASYNCcnt) ---
// ROCm 7.2 has no load-to-lds builtin for gfx1250; use the ISA mnemonic
// directly (bridge doc: inline asm is the portable path for the async ops).
// LDS byte offset = low 32 bits of the flat shared pointer (ISA 10.2 aperture).
#if defined(__HIP_DEVICE_COMPILE__)
#define ASYNC_COPY_B128(gptr, ldsptr)                                          \
  asm volatile("global_load_async_to_lds_b128 %0, %1, off"                     \
               :: "v"((unsigned)(size_t)(ldsptr)), "v"((const void*)(gptr))    \
               : "memory")
#define ASYNC_WAIT0() asm volatile("s_wait_asynccnt 0x0" ::: "memory")
#else
#define ASYNC_COPY_B128(gptr, ldsptr) do { (void)(gptr); (void)(ldsptr); } while (0)
#define ASYNC_WAIT0() do { } while (0)
#endif

__device__ __forceinline__ unsigned short f32_to_bf16_rn(float f) {
  unsigned int u = __float_as_uint(f);
  unsigned int lsb = (u >> 16) & 1u;
  u += 0x7fffu + lsb;                       // round to nearest even
  return (unsigned short)(u >> 16);
}

// ---------------------------------------------------------------------------
// f32 -> bf16 (row-major preserved), 4 elements per thread
// ---------------------------------------------------------------------------
__global__ void cvt_f32_bf16_kernel(const float4* __restrict__ src,
                                    ushort4* __restrict__ dst, int n4) {
  int i = blockIdx.x * blockDim.x + threadIdx.x;
  if (i < n4) {
    float4 f = src[i];
    ushort4 o;
    o.x = f32_to_bf16_rn(f.x);
    o.y = f32_to_bf16_rn(f.y);
    o.z = f32_to_bf16_rn(f.z);
    o.w = f32_to_bf16_rn(f.w);
    dst[i] = o;
  }
}

// ---------------------------------------------------------------------------
// f32 [R][C] -> bf16 transposed [C][R]; 32x32 LDS tile, both sides coalesced.
// ---------------------------------------------------------------------------
__global__ void __launch_bounds__(256)
cvtT_f32_bf16_kernel(const float* __restrict__ src,
                     unsigned short* __restrict__ dst, int R, int C) {
  __shared__ float tile[32][33];
  const int bx = blockIdx.x * 32;   // col base in src
  const int by = blockIdx.y * 32;   // row base in src
  const int tx = threadIdx.x;       // 0..31
  const int ty = threadIdx.y;       // 0..7
#pragma unroll
  for (int i = 0; i < 32; i += 8)
    tile[ty + i][tx] = src[(size_t)(by + ty + i) * C + (bx + tx)];
  __syncthreads();
#pragma unroll
  for (int i = 0; i < 32; i += 8)
    dst[(size_t)(bx + ty + i) * R + (by + tx)] = f32_to_bf16_rn(tile[tx][ty + i]);
}

// ---------------------------------------------------------------------------
// C[M x Nc] (f32) = A[M x K] (bf16, row-major) * Bt[Nc x K] (bf16, B transposed)
// block = 256 threads = 8 wave32; block tile 64(M) x 32(N); K-step 32.
// Double-buffered ASYNC (ASYNCcnt) LDS staging; fragments are b128 LDS loads.
// epi: 0=none, 1=ELU, 2=ReLU
// ---------------------------------------------------------------------------
__global__ void __launch_bounds__(256)
wmma_gemm_bf16_kernel(const unsigned short* __restrict__ A,
                      const unsigned short* __restrict__ Bt,
                      float* __restrict__ C,
                      int M, int Nc, int K, int epi) {
  __shared__ __align__(16) unsigned short ldsA[2][64][40];  // 80B pitch
  __shared__ __align__(16) unsigned short ldsB[2][32][40];

  const int t    = threadIdx.x;
  const int lane = t & 31;
  const int wave = t >> 5;
  const int mi   = wave >> 1;          // 0..3
  const int ni   = wave & 1;           // 0..1
  const int m0   = blockIdx.x * 64;
  const int n0   = blockIdx.y * 32;

  const int ar  = t >> 2;              // 0..63  A tile row
  const int ac  = (t & 3) * 8;         // 0/8/16/24
  const int br  = (t & 127) >> 2;      // 0..31  Bt tile row (= B column)
  const int bc2 = ((t & 127) & 3) * 8;

  const unsigned short* gA = A  + (size_t)(m0 + ar) * K + ac;
  const unsigned short* gB = Bt + (size_t)(n0 + br) * K + bc2;

  v8f acc = {0.f, 0.f, 0.f, 0.f, 0.f, 0.f, 0.f, 0.f};

  const int arow = mi * 16 + (lane & 15);
  const int bcol = ni * 16 + (lane & 15);
  const int koff = (lane >> 4) * 8;    // wave32 hi/lo half-K split

  const int T = K / 32;

  auto stage = [&](int i, int buf) {
    // fire-and-forget DMA: global -> LDS with no VGPR round-trip
    ASYNC_COPY_B128(gA + (size_t)i * 32, &ldsA[buf][ar][ac]);
    if (t < 128)
      ASYNC_COPY_B128(gB + (size_t)i * 32, &ldsB[buf][br][bc2]);
  };

  stage(0, 0);
  for (int i = 0; i < T; ++i) {
    ASYNC_WAIT0();                     // my DMA into tile i has landed in LDS
    __syncthreads();                   // everyone's tile i visible; i-1 reads done
    if (i + 1 < T) {
      stage(i + 1, (i + 1) & 1);       // overlap DMA of tile i+1 with WMMA of tile i
      if (i + 2 < T) __builtin_prefetch(gA + (size_t)(i + 2) * 32, 0, 3);
    }
    const int b = i & 1;
    // A fragment: K koff..koff+7 and 16+koff..23+koff  (contiguous b128 pairs)
    U32x8 ua, ub;
    ua.lo = *(const uint4*)&ldsA[b][arow][koff];
    ua.hi = *(const uint4*)&ldsA[b][arow][16 + koff];
    // B fragment: column bcol of B == row bcol of Bt, same K striping
    ub.lo = *(const uint4*)&ldsB[b][bcol][koff];
    ub.hi = *(const uint4*)&ldsB[b][bcol][16 + koff];
    v16bf af  = __builtin_bit_cast(v16bf, ua);
    v16bf bfv = __builtin_bit_cast(v16bf, ub);
    acc = __builtin_amdgcn_wmma_f32_16x16x32_bf16(
        /*neg_a=*/false, af, /*neg_b=*/false, bfv,
        /*c_mod=*/(short)0, acc, /*reuse_a=*/false, /*reuse_b=*/false);
  }

  // store 16x16 f32 tile: VGPR v -> row v + 8*(lane>>4), col lane&15
  const int crow0 = m0 + mi * 16 + 8 * (lane >> 4);
  const int ccol  = n0 + ni * 16 + (lane & 15);
#pragma unroll
  for (int v = 0; v < 8; ++v) {
    float x = acc[v];
    if (epi == 1)      x = (x > 0.f) ? x : (__expf(x) - 1.f);  // ELU
    else if (epi == 2) x = fmaxf(x, 0.f);                      // ReLU (= relu∘elu)
    C[(size_t)(crow0 + v) * Nc + ccol] = x;
  }
}

// ---------------------------------------------------------------------------
// f_src[i] = Wh[i,:] . a[0:D] ;  f_dst[i] = Wh[i,:] . a[D:2D]
// ---------------------------------------------------------------------------
__global__ void __launch_bounds__(256)
fvec_kernel(const float* __restrict__ Wh, const float* __restrict__ a,
            float* __restrict__ fs, float* __restrict__ fd) {
  __shared__ float r1[256], r2[256];
  const int row = blockIdx.x, t = threadIdx.x;
  float w = Wh[(size_t)row * DV + t];
  r1[t] = w * a[t];
  r2[t] = w * a[DV + t];
  __syncthreads();
  for (int s = 128; s > 0; s >>= 1) {
    if (t < s) { r1[t] += r1[t + s]; r2[t] += r2[t + s]; }
    __syncthreads();
  }
  if (t == 0) { fs[row] = r1[0]; fd[row] = r2[0]; }
}

// ---------------------------------------------------------------------------
// attn[row,:] = softmax( mask(adj>0, leaky_relu(fs[row]+fd[:])) ), bf16 out.
// One block per row; 16KB e-row in LDS (320KB/WGP on CDNA5).
// ---------------------------------------------------------------------------
__global__ void __launch_bounds__(256)
attn_softmax_kernel(const float* __restrict__ adj,
                    const float* __restrict__ fs,
                    const float* __restrict__ fd,
                    unsigned short* __restrict__ attn) {
  __shared__ float erow[NV];
  __shared__ float red[256];
  const int row = blockIdx.x, t = threadIdx.x;
  const float fsr = fs[row];

  float lmax = -INFINITY;
  for (int j = t; j < NV; j += 256) {
    float aj = adj[(size_t)row * NV + j];
    float e;
    if (aj > 0.f) {
      float x = fsr + fd[j];
      e = (x > 0.f) ? x : LRELU_ALPHA * x;
    } else {
      e = -INFINITY;                  // reference uses -9e15; identical after softmax
    }
    erow[j] = e;
    lmax = fmaxf(lmax, e);
  }
  red[t] = lmax; __syncthreads();
  for (int s = 128; s > 0; s >>= 1) {
    if (t < s) red[t] = fmaxf(red[t], red[t + s]);
    __syncthreads();
  }
  const float m = red[0];             // finite: self-loop guarantees a valid entry
  __syncthreads();

  float lsum = 0.f;
  for (int j = t; j < NV; j += 256) {
    float ex = __expf(erow[j] - m);   // exp(-inf) == 0 for masked entries
    erow[j] = ex;
    lsum += ex;
  }
  red[t] = lsum; __syncthreads();
  for (int s = 128; s > 0; s >>= 1) {
    if (t < s) red[t] += red[t + s];
    __syncthreads();
  }
  const float inv = 1.f / red[0];
  for (int j = t; j < NV; j += 256)
    attn[(size_t)row * NV + j] = f32_to_bf16_rn(erow[j] * inv);
}

// ---------------------------------------------------------------------------
extern "C" void kernel_launch(void* const* d_in, const int* in_sizes, int n_in,
                              void* d_out, int out_size, void* d_ws, size_t ws_size,
                              hipStream_t stream) {
  (void)in_sizes; (void)n_in; (void)out_size; (void)ws_size;

  const float* x   = (const float*)d_in[0];
  const float* adj = (const float*)d_in[1];
  const float* W11 = (const float*)d_in[2];
  const float* a11 = (const float*)d_in[3];
  const float* W12 = (const float*)d_in[4];
  const float* a12 = (const float*)d_in[5];
  const float* W21 = (const float*)d_in[6];
  const float* a21 = (const float*)d_in[7];
  const float* W22 = (const float*)d_in[8];
  const float* a22 = (const float*)d_in[9];

  char* ws = (char*)d_ws;
  size_t off = 0;
  auto carve = [&](size_t bytes) {
    off = (off + 255) & ~(size_t)255;
    void* p = ws + off;
    off += bytes;
    return p;
  };
  // ab: bf16(X or h) for GEMM-1 A, then overwritten with bf16 attn (stream-ordered reuse)
  unsigned short* ab  = (unsigned short*)carve((size_t)NV * NV * 2);  // 32 MB
  // wbT: bf16 transposed B: W^T for GEMM-1, then Wh^T for GEMM-2
  unsigned short* wbT = (unsigned short*)carve((size_t)NV * DV * 2);  //  2 MB
  float* whf = (float*)carve((size_t)NV * DV * 4);                    //  4 MB
  float* hf  = (float*)carve((size_t)NV * DV * 4);                    //  4 MB
  float* fsv = (float*)carve((size_t)NV * 4);
  float* fdv = (float*)carve((size_t)NV * 4);

  auto cvt = [&](const float* s, unsigned short* d, int n) {
    int n4 = n / 4;
    cvt_f32_bf16_kernel<<<(n4 + 255) / 256, 256, 0, stream>>>(
        (const float4*)s, (ushort4*)d, n4);
  };
  auto cvtT = [&](const float* s, unsigned short* d, int R, int Ccols) {
    cvtT_f32_bf16_kernel<<<dim3(Ccols / 32, R / 32), dim3(32, 8), 0, stream>>>(s, d, R, Ccols);
  };

  // One GAT attention layer:  h_out = epi( softmax(mask(lrelu(f+f^T))) @ (h_in @ W) )
  auto layer = [&](const float* h_in, int Kin, const float* W, const float* a,
                   float* h_out, int epi) {
    cvt(h_in, ab, NV * Kin);                                   // A (row-major bf16)
    cvtT(W, wbT, Kin, DV);                                     // W^T: [DV][Kin]
    wmma_gemm_bf16_kernel<<<dim3(NV / 64, DV / 32), 256, 0, stream>>>(
        ab, wbT, whf, NV, DV, Kin, /*epi=*/0);                 // Wh = h @ W
    fvec_kernel<<<NV, 256, 0, stream>>>(whf, a, fsv, fdv);     // f_src, f_dst
    cvtT(whf, wbT, NV, DV);                                    // Wh^T: [DV][NV]
    attn_softmax_kernel<<<NV, 256, 0, stream>>>(adj, fsv, fdv, ab);
    wmma_gemm_bf16_kernel<<<dim3(NV / 64, DV / 32), 256, 0, stream>>>(
        ab, wbT, h_out, NV, DV, NV, epi);                      // epi(attn @ Wh)
  };

  layer(x,  NV, W11, a11, hf, /*ELU*/ 1);
  layer(hf, DV, W12, a12, hf, /*ReLU == relu∘elu*/ 2);
  layer(hf, DV, W21, a21, hf, /*ELU*/ 1);
  layer(hf, DV, W22, a22, (float*)d_out, /*ELU*/ 1);
}
